// DistrDeepAR_7919919694317
// MI455X (gfx1250) — matrix operand order repaired and checked by reading, not compiled, not run on hardware
//
#include <hip/hip_runtime.h>

// ---------------- problem constants ----------------
#define B_      256
#define L_      412
#define C_      3
#define CTX_    336
#define HOR_    48
#define NS_     100
#define HID_    64
#define IND_    15
#define MAXLAG_ 28
#define T_      384        // L_ - MAXLAG_
#define TM1_    383        // T_ - 1
#define G_      256        // 4*HID_

__constant__ int d_LAG[10] = {1, 2, 3, 4, 5, 6, 7, 14, 21, 28};

// ---------------- WMMA types / helpers ----------------
typedef __attribute__((ext_vector_type(16))) _Float16 v16h;
typedef __attribute__((ext_vector_type(8)))  _Float16 v8h;
typedef __attribute__((ext_vector_type(8)))  float    v8f;

__device__ __forceinline__ v8f wmma16(v16h a, v16h b, v8f c) {
    // D = A(16x32 f16) * B(32x16 f16) + C(16x16 f32)
    return __builtin_amdgcn_wmma_f32_16x16x32_f16(
        /*neg_a=*/false, a, /*neg_b=*/false, b,
        /*c_mod=*/(short)0, c, /*reuse_a=*/false, /*reuse_b=*/false);
}

// B operand (32x16 f16, K-major per lane) from global weights stored [N][K] row-major.
__device__ __forceinline__ v16h load_b16(const _Float16* __restrict__ W, int SK,
                                         int ntile, int ktile, int lane) {
    int n = ntile * 16 + (lane & 15);
    int k = ktile * 32 + ((lane >> 4) << 4);
    return *(const v16h*)(W + (size_t)n * SK + k);
}

// A operand (16x32 f16) from row-major LDS [16][SK].
__device__ __forceinline__ v16h load_a16(const _Float16* __restrict__ S, int SK,
                                         int ktile, int lane) {
    int m  = lane & 15;
    int kb = ktile * 32 + ((lane >> 4) << 4);
    const _Float16* p = S + m * SK + kb;
    return *(const v16h*)(p);
}

// ---------------- fast transcendentals (single-instruction gfx1250 paths) ----------
__device__ __forceinline__ float fast_rcpf(float x) { return __builtin_amdgcn_rcpf(x); }
__device__ __forceinline__ float fast_sigmoidf(float x) {
    return fast_rcpf(1.f + __expf(-x));            // v_exp + v_rcp
}
__device__ __forceinline__ float fast_tanhf(float x) {
#if __has_builtin(__builtin_amdgcn_tanhf)
    return __builtin_amdgcn_tanhf(x);              // v_tanh_f32
#else
    return 1.f - 2.f * fast_rcpf(1.f + __expf(2.f * x));
#endif
}
__device__ __forceinline__ float softplusf_(float x) {
    return (x > 20.f) ? x : __logf(1.f + __expf(x));
}

// ---------------- counter-based RNG (deterministic across replays) ----------------
__device__ __forceinline__ unsigned pcg_next(unsigned& st) {
    unsigned x = st;
    st = st * 747796405u + 2891336453u;
    x = ((x >> ((x >> 28u) + 4u)) ^ x) * 277803737u;
    return (x >> 22u) ^ x;
}
__device__ __forceinline__ unsigned mkseed(int q, int t) {
    unsigned s = (unsigned)q * 2654435761u ^ ((unsigned)t * 0x9E3779B9u) ^ 0x85EBCA6Bu;
    s = s * 747796405u + 2891336453u;
    return s;
}
__device__ __forceinline__ float rng_u01(unsigned& s) {
    return (pcg_next(s) >> 8) * (1.f / 16777216.f) + 1e-8f;
}
__device__ __forceinline__ float rng_normal(unsigned& s) {
    float u1 = rng_u01(s), u2 = rng_u01(s);
    return sqrtf(-2.f * __logf(u1)) * __cosf(6.28318530718f * u2);
}
__device__ __forceinline__ float rng_gamma(unsigned& s, float a) {
    // Marsaglia-Tsang, a >= 1 (a = df/2, df >= 2)
    float d = a - (1.f / 3.f);
    float c = rsqrtf(9.f * d);
    for (int it = 0; it < 8; ++it) {
        float x = rng_normal(s);
        float t = 1.f + c * x;
        if (t <= 0.f) continue;
        float v = t * t * t;
        float u = rng_u01(s);
        if (__logf(u) < 0.5f * x * x + d - d * v + d * __logf(v)) return d * v;
    }
    return d;
}
__device__ __forceinline__ float t_noise(unsigned& s, float df) {
    float z = rng_normal(s);
    float g = rng_gamma(s, 0.5f * df);
    return z * sqrtf(df / (2.f * g));
}

// ---------------- fused LSTM layer step (per 256-thread WG, M=16 batch tile) ----------------
template <int NKX>
__device__ __forceinline__ void lstm_layer_step(
    const _Float16* __restrict__ sX, int sxStride,
    const _Float16* __restrict__ sHin,
    const v16h (&bx)[2][NKX], const v16h (&bh)[2][2],
    const float (&bias)[2],
    float* __restrict__ sG, float* __restrict__ sC, _Float16* __restrict__ sHout,
    int tid)
{
    int wave = tid >> 5, lane = tid & 31;
    v16h ax[NKX];
#pragma unroll
    for (int kt = 0; kt < NKX; ++kt) ax[kt] = load_a16(sX, sxStride, kt, lane);
    v16h ah0 = load_a16(sHin, 64, 0, lane);
    v16h ah1 = load_a16(sHin, 64, 1, lane);

#pragma unroll
    for (int j = 0; j < 2; ++j) {
        v8f acc;
#pragma unroll
        for (int r = 0; r < 8; ++r) acc[r] = bias[j];
#pragma unroll
        for (int kt = 0; kt < NKX; ++kt) acc = wmma16(ax[kt], bx[j][kt], acc);
        acc = wmma16(ah0, bh[j][0], acc);
        acc = wmma16(ah1, bh[j][1], acc);

        int n    = (wave * 2 + j) * 16 + (lane & 15);
        int nc   = n & 63;
        // gate id (0:i 1:f 2:g 3:o) is wave-uniform -> scalar branch, no EXEC juggling
        int gs   = __builtin_amdgcn_readfirstlane(n >> 6);
        float* gb = sG + gs * 1024 + nc;
        int   mb  = (lane >> 4) << 3;
        if (gs == 2) {
#pragma unroll
            for (int r = 0; r < 8; ++r) gb[(mb + r) * 64] = fast_tanhf(acc[r]);
        } else {
#pragma unroll
            for (int r = 0; r < 8; ++r) gb[(mb + r) * 64] = fast_sigmoidf(acc[r]);
        }
    }
    __syncthreads();
#pragma unroll
    for (int u = 0; u < 4; ++u) {
        int   e  = tid * 4 + u;           // e = m*64 + n
        float i_ = sG[e], f_ = sG[1024 + e], g_ = sG[2048 + e], o_ = sG[3072 + e];
        float c  = f_ * sC[e] + i_ * g_;
        sC[e]    = c;
        sHout[e] = (_Float16)(o_ * fast_tanhf(c));
    }
    __syncthreads();
}

// ---------------- kernel 1: tscale + feature build (f16, padded to 16) ----------------
__global__ void k_prep(const float* __restrict__ X, const unsigned char* __restrict__ mask,
                       const float* __restrict__ emb,
                       float* __restrict__ tscale, float* __restrict__ buf0,
                       _Float16* __restrict__ feat)
{
    int b = blockIdx.x, tid = threadIdx.x;
    __shared__ float red[256];
    float sa = 0.f, sm = 0.f;
    for (int t = tid; t < CTX_; t += 256) {
        int   gt = MAXLAG_ + t;
        float m  = mask[b * L_ + gt] ? 1.f : 0.f;
        sa += fabsf(X[((size_t)b * L_ + gt) * C_ + 0]) * m;
        sm += m;
    }
    red[tid] = sa; __syncthreads();
    for (int o = 128; o > 0; o >>= 1) { if (tid < o) red[tid] += red[tid + o]; __syncthreads(); }
    float tot_a = red[0]; __syncthreads();
    red[tid] = sm; __syncthreads();
    for (int o = 128; o > 0; o >>= 1) { if (tid < o) red[tid] += red[tid + o]; __syncthreads(); }
    float tot_m = red[0];

    float ts = fmaxf(tot_a / fmaxf(tot_m, 1.f), 1e-10f);
    if (tid == 0) tscale[b] = ts;
    float inv = 1.f / ts, ls = logf(ts), e = emb[0];

    for (int i = tid; i < MAXLAG_; i += 256)
        buf0[b * MAXLAG_ + i] = X[((size_t)b * L_ + (L_ - HOR_ - 1 - i)) * C_ + 0] * inv;

    for (int t = tid; t < TM1_; t += 256) {
        int gt = MAXLAG_ + t;
        const float* xr = X + ((size_t)b * L_ + gt) * C_;
        float x0 = (gt >= L_ - HOR_) ? 0.f : xr[0] * inv;
        _Float16 row[16];
        row[0] = (_Float16)x0;
#pragma unroll
        for (int j = 0; j < 10; ++j) {
            float lv = 0.f;
            if (t < CTX_) lv = X[((size_t)b * L_ + (gt - d_LAG[j])) * C_ + 0] * inv;
            row[1 + j] = (_Float16)lv;
        }
        row[11] = (_Float16)e;
        row[12] = (_Float16)xr[2];
        row[13] = (_Float16)ls;
        row[14] = (_Float16)xr[1];
        row[15] = (_Float16)0.f;
        _Float16* dst = feat + ((size_t)b * TM1_ + t) * 16;
        *(v8h*)(dst)     = *(const v8h*)(row);
        *(v8h*)(dst + 8) = *(const v8h*)(row + 8);
    }
}

// ---------------- kernel 2: weight conversion to f16 [N][K] + bias fusion ----------------
__global__ void k_wcvt(const float* __restrict__ Wih0, const float* __restrict__ Whh0,
                       const float* __restrict__ Wih1, const float* __restrict__ Whh1,
                       const float* __restrict__ bih0, const float* __restrict__ bhh0,
                       const float* __restrict__ bih1, const float* __restrict__ bhh1,
                       _Float16* __restrict__ Wih0h, _Float16* __restrict__ Whh0h,
                       _Float16* __restrict__ Wih1h, _Float16* __restrict__ Whh1h,
                       float* __restrict__ bias0, float* __restrict__ bias1)
{
    int n = blockIdx.x, k = threadIdx.x;           // grid 256 x 64
    if (k < 32) Wih0h[n * 32 + k] = (_Float16)((k < IND_) ? Wih0[n * IND_ + k] : 0.f);
    Whh0h[n * 64 + k] = (_Float16)Whh0[n * 64 + k];
    Wih1h[n * 64 + k] = (_Float16)Wih1[n * 64 + k];
    Whh1h[n * 64 + k] = (_Float16)Whh1[n * 64 + k];
    if (k == 0) { bias0[n] = bih0[n] + bhh0[n]; bias1[n] = bih1[n] + bhh1[n]; }
}

// ---------------- kernel 3: context LSTM (16 blocks, 336 steps, WMMA) ----------------
__global__ void __launch_bounds__(256, 1) k_ctx(
    const _Float16* __restrict__ feat,
    const _Float16* __restrict__ Wih0h, const _Float16* __restrict__ Whh0h,
    const _Float16* __restrict__ Wih1h, const _Float16* __restrict__ Whh1h,
    const float* __restrict__ bias0, const float* __restrict__ bias1,
    const float* __restrict__ w_df, const float* __restrict__ b_df,
    const float* __restrict__ w_loc, const float* __restrict__ b_loc,
    const float* __restrict__ w_sc, const float* __restrict__ b_sc,
    float* __restrict__ dls_df, float* __restrict__ dls_loc, float* __restrict__ dls_sc,
    float* __restrict__ st_h0, float* __restrict__ st_c0,
    float* __restrict__ st_h1, float* __restrict__ st_c1)
{
    __shared__ __align__(32) _Float16 sAx[16 * 32];
    __shared__ __align__(32) _Float16 sH0[1024];
    __shared__ __align__(32) _Float16 sH1[1024];
    __shared__ float sC0[1024], sC1[1024];
    __shared__ float sG[4 * 1024];
    __shared__ float swdf[64], swloc[64], swsc[64];

    int tid = threadIdx.x, wave = tid >> 5, lane = tid & 31;
    int b0  = blockIdx.x * 16;

    // resident weights (B-operand layout) in VGPRs
    v16h bx0[2][1], bh0[2][2], bx1[2][2], bh1[2][2];
    float bv0[2], bv1[2];
#pragma unroll
    for (int j = 0; j < 2; ++j) {
        int nt = wave * 2 + j;
        bx0[j][0] = load_b16(Wih0h, 32, nt, 0, lane);
#pragma unroll
        for (int kt = 0; kt < 2; ++kt) {
            bh0[j][kt] = load_b16(Whh0h, 64, nt, kt, lane);
            bx1[j][kt] = load_b16(Wih1h, 64, nt, kt, lane);
            bh1[j][kt] = load_b16(Whh1h, 64, nt, kt, lane);
        }
        bv0[j] = bias0[nt * 16 + (lane & 15)];
        bv1[j] = bias1[nt * 16 + (lane & 15)];
    }

    for (int e = tid; e < 1024; e += 256) {
        sH0[e] = (_Float16)0.f; sH1[e] = (_Float16)0.f; sC0[e] = 0.f; sC1[e] = 0.f;
    }
    { int m = tid >> 4, k = tid & 15; sAx[m * 32 + 16 + k] = (_Float16)0.f; } // K-pad
    if (tid < 64) { swdf[tid] = w_df[tid]; swloc[tid] = w_loc[tid]; swsc[tid] = w_sc[tid]; }
    __syncthreads();

    float bdf = b_df[0], bloc = b_loc[0], bsc = b_sc[0];

    // double-buffered feature rows: one 32B row per sequence, prefetched a step ahead
    v8h f0 = {}, f1 = {};
    if (tid < 16) {
        const _Float16* p = feat + ((size_t)(b0 + tid) * TM1_) * 16;
        f0 = *(const v8h*)(p); f1 = *(const v8h*)(p + 8);
    }

    for (int t = 0; t < CTX_; ++t) {
        if (tid < 16) {
            *(v8h*)(sAx + tid * 32)     = f0;      // stage step t
            *(v8h*)(sAx + tid * 32 + 8) = f1;
            int tn = (t + 1 < CTX_) ? t + 1 : t;   // prefetch step t+1 (hidden by WMMA)
            const _Float16* p = feat + ((size_t)(b0 + tid) * TM1_ + tn) * 16;
            f0 = *(const v8h*)(p); f1 = *(const v8h*)(p + 8);
        }
        __syncthreads();
        lstm_layer_step<1>(sAx, 32, sH0, bx0, bh0, bv0, sG, sC0, sH0, tid);
        lstm_layer_step<2>(sH0, 64, sH1, bx1, bh1, bv1, sG, sC1, sH1, tid);
        if (tid < 16) {
            int m = tid; float a0 = 0.f, a1 = 0.f, a2 = 0.f;
#pragma unroll 8
            for (int n = 0; n < 64; ++n) {
                float h = (float)sH1[m * 64 + n];
                a0 += h * swdf[n]; a1 += h * swloc[n]; a2 += h * swsc[n];
            }
            int gb = b0 + m;
            dls_df [gb * CTX_ + t] = 2.f + softplusf_(a0 + bdf);
            dls_loc[gb * CTX_ + t] = a1 + bloc;
            dls_sc [gb * CTX_ + t] = softplusf_(a2 + bsc);
        }
        __syncthreads();
    }
    for (int e = tid; e < 1024; e += 256) {
        int m = e >> 6, n = e & 63, gb = b0 + m;
        st_h0[gb * 64 + n] = (float)sH0[e]; st_c0[gb * 64 + n] = sC0[e];
        st_h1[gb * 64 + n] = (float)sH1[e]; st_c1[gb * 64 + n] = sC1[e];
    }
}

// ---------------- kernel 4: autoregressive sampling (1600 blocks x 47 steps) ----------------
__global__ void __launch_bounds__(256, 1) k_hor(
    const _Float16* __restrict__ feat,
    const _Float16* __restrict__ Wih0h, const _Float16* __restrict__ Whh0h,
    const _Float16* __restrict__ Wih1h, const _Float16* __restrict__ Whh1h,
    const float* __restrict__ bias0, const float* __restrict__ bias1,
    const float* __restrict__ w_df, const float* __restrict__ b_df,
    const float* __restrict__ w_loc, const float* __restrict__ b_loc,
    const float* __restrict__ w_sc, const float* __restrict__ b_sc,
    const float* __restrict__ dls_df, const float* __restrict__ dls_loc,
    const float* __restrict__ dls_sc,
    const float* __restrict__ st_h0, const float* __restrict__ st_c0,
    const float* __restrict__ st_h1, const float* __restrict__ st_c1,
    const float* __restrict__ tscale, const float* __restrict__ buf0,
    float* __restrict__ ysamp)
{
    __shared__ __align__(32) _Float16 sAx[16 * 32];
    __shared__ __align__(32) _Float16 sH0[1024];
    __shared__ __align__(32) _Float16 sH1[1024];
    __shared__ float sC0[1024], sC1[1024];
    __shared__ float sG[4 * 1024];
    __shared__ float lagb[16][MAXLAG_];
    __shared__ float prevv[16], tsq[16];
    __shared__ float swdf[64], swloc[64], swsc[64];

    int tid = threadIdx.x, wave = tid >> 5, lane = tid & 31;
    int q0  = blockIdx.x * 16;

    v16h bx0[2][1], bh0[2][2], bx1[2][2], bh1[2][2];
    float bv0[2], bv1[2];
#pragma unroll
    for (int j = 0; j < 2; ++j) {
        int nt = wave * 2 + j;
        bx0[j][0] = load_b16(Wih0h, 32, nt, 0, lane);
#pragma unroll
        for (int kt = 0; kt < 2; ++kt) {
            bh0[j][kt] = load_b16(Whh0h, 64, nt, kt, lane);
            bx1[j][kt] = load_b16(Wih1h, 64, nt, kt, lane);
            bh1[j][kt] = load_b16(Whh1h, 64, nt, kt, lane);
        }
        bv0[j] = bias0[nt * 16 + (lane & 15)];
        bv1[j] = bias1[nt * 16 + (lane & 15)];
    }

    for (int e = tid; e < 1024; e += 256) {
        int m = e >> 6, n = e & 63, q = q0 + m, gb = q / NS_;
        sH0[e] = (_Float16)st_h0[gb * 64 + n]; sC0[e] = st_c0[gb * 64 + n];
        sH1[e] = (_Float16)st_h1[gb * 64 + n]; sC1[e] = st_c1[gb * 64 + n];
    }
    { int m = tid >> 4, k = tid & 15; sAx[m * 32 + 16 + k] = (_Float16)0.f; }
    if (tid < 64) { swdf[tid] = w_df[tid]; swloc[tid] = w_loc[tid]; swsc[tid] = w_sc[tid]; }
    if (tid < 16) {
        int m = tid, q = q0 + m, gb = q / NS_;
        tsq[m] = tscale[gb];
        for (int i = 0; i < MAXLAG_; ++i) lagb[m][i] = buf0[gb * MAXLAG_ + i];
        unsigned rs = mkseed(q, CTX_ - 1);
        float df = dls_df[gb * CTX_ + CTX_ - 1];
        float lo = dls_loc[gb * CTX_ + CTX_ - 1];
        float sc = dls_sc[gb * CTX_ + CTX_ - 1];
        prevv[m] = (lo + sc * t_noise(rs, df)) * tsq[m];
    }
    __syncthreads();

    float bdf = b_df[0], bloc = b_loc[0], bsc = b_sc[0];

    for (int step = 0; step < HOR_ - 1; ++step) {
        int tt = CTX_ + step;
        if (tid < 16) {
            int m = tid, q = q0 + m, gb = q / NS_;
            float ns = prevv[m] / tsq[m];
            const _Float16* fr = feat + ((size_t)gb * TM1_ + tt) * 16;
            // prefetch next step's feature row into caches (global_prefetch_b8)
            int tn = (tt + 1 < TM1_) ? tt + 1 : tt;
            __builtin_prefetch(feat + ((size_t)gb * TM1_ + tn) * 16, 0, 1);
            sAx[m * 32 + 0] = (_Float16)ns;
#pragma unroll
            for (int j = 0; j < 10; ++j)
                sAx[m * 32 + 1 + j] = (_Float16)lagb[m][d_LAG[j] - 1];
            sAx[m * 32 + 11] = fr[11]; sAx[m * 32 + 12] = fr[12];
            sAx[m * 32 + 13] = fr[13]; sAx[m * 32 + 14] = fr[14];
            sAx[m * 32 + 15] = (_Float16)0.f;
            for (int i = MAXLAG_ - 1; i > 0; --i) lagb[m][i] = lagb[m][i - 1];
            lagb[m][0] = ns;
        }
        __syncthreads();
        lstm_layer_step<1>(sAx, 32, sH0, bx0, bh0, bv0, sG, sC0, sH0, tid);
        lstm_layer_step<2>(sH0, 64, sH1, bx1, bh1, bv1, sG, sC1, sH1, tid);
        if (tid < 16) {
            int m = tid, q = q0 + m, gb = q / NS_, ss = q % NS_;
            float a0 = 0.f, a1 = 0.f, a2 = 0.f;
#pragma unroll 8
            for (int n = 0; n < 64; ++n) {
                float h = (float)sH1[m * 64 + n];
                a0 += h * swdf[n]; a1 += h * swloc[n]; a2 += h * swsc[n];
            }
            float df = 2.f + softplusf_(a0 + bdf);
            float lo = a1 + bloc;
            float sc = softplusf_(a2 + bsc);
            unsigned rs = mkseed(q, tt);
            float y = (lo + sc * t_noise(rs, df)) * tsq[m];
            prevv[m] = y;
            ysamp[((size_t)gb * NS_ + ss) * (HOR_ - 1) + step] = y;
        }
        __syncthreads();
    }
}

// ---------------- kernel 5: median over NS samples (sample buffer in LDS) -------------
// grid (TM1_, 2) x 128 threads; 100x128 f32 = 50KB LDS per block.
// Layout vbuf[s][thread]: consecutive lanes hit consecutive banks -> conflict-free ds ops.
__global__ void __launch_bounds__(128, 2) k_median(
    const float* __restrict__ dls_df, const float* __restrict__ dls_loc,
    const float* __restrict__ dls_sc, const float* __restrict__ tscale,
    const float* __restrict__ ysamp, float* __restrict__ out)
{
    __shared__ float vbuf[NS_][128];
    int t   = blockIdx.x;                       // 0..382
    int b   = blockIdx.y * 128 + threadIdx.x;   // 0..255
    int lid = threadIdx.x;

    if (t < CTX_) {
        float df = dls_df[b * CTX_ + t], lo = dls_loc[b * CTX_ + t];
        float sc = dls_sc[b * CTX_ + t], ts = tscale[b];
        for (int s = 0; s < NS_; ++s) {
            unsigned rs = mkseed(b * NS_ + s, t);
            vbuf[s][lid] = (lo + sc * t_noise(rs, df)) * ts;
        }
    } else {
        for (int s = 0; s < NS_; ++s)
            vbuf[s][lid] = ysamp[((size_t)b * NS_ + s) * (HOR_ - 1) + (t - CTX_)];
    }
    // partial selection sort (per-thread column) to index 49 = (NS-1)/2
    for (int i = 0; i <= 49; ++i) {
        int   mi = i;
        float mv = vbuf[i][lid];
        for (int j = i + 1; j < NS_; ++j) {
            float vj = vbuf[j][lid];
            if (vj < mv) { mv = vj; mi = j; }
        }
        vbuf[mi][lid] = vbuf[i][lid];
        vbuf[i][lid]  = mv;
    }
    out[b * TM1_ + t] = vbuf[49][lid];
}

// ---------------- host launch ----------------
extern "C" void kernel_launch(void* const* d_in, const int* in_sizes, int n_in,
                              void* d_out, int out_size, void* d_ws, size_t ws_size,
                              hipStream_t stream)
{
    (void)in_sizes; (void)n_in; (void)out_size; (void)ws_size;
    const float*         X     = (const float*)d_in[0];
    const unsigned char* mask  = (const unsigned char*)d_in[1];
    const float* Wih0 = (const float*)d_in[2];
    const float* Whh0 = (const float*)d_in[3];
    const float* bih0 = (const float*)d_in[4];
    const float* bhh0 = (const float*)d_in[5];
    const float* Wih1 = (const float*)d_in[6];
    const float* Whh1 = (const float*)d_in[7];
    const float* bih1 = (const float*)d_in[8];
    const float* bhh1 = (const float*)d_in[9];
    const float* w_df = (const float*)d_in[10];
    const float* b_df = (const float*)d_in[11];
    const float* w_lo = (const float*)d_in[12];
    const float* b_lo = (const float*)d_in[13];
    const float* w_sc = (const float*)d_in[14];
    const float* b_sc = (const float*)d_in[15];
    const float* emb  = (const float*)d_in[16];

    char*  ws  = (char*)d_ws;
    size_t off = 0;
    auto alloc = [&](size_t bytes) -> void* {
        void* p = ws + off;
        off = (off + bytes + 255) & ~(size_t)255;
        return p;
    };
    float*    tscale = (float*)alloc(B_ * 4);
    float*    buf0   = (float*)alloc(B_ * MAXLAG_ * 4);
    _Float16* feat   = (_Float16*)alloc((size_t)B_ * TM1_ * 16 * 2);
    _Float16* Wih0h  = (_Float16*)alloc(G_ * 32 * 2);
    _Float16* Whh0h  = (_Float16*)alloc(G_ * 64 * 2);
    _Float16* Wih1h  = (_Float16*)alloc(G_ * 64 * 2);
    _Float16* Whh1h  = (_Float16*)alloc(G_ * 64 * 2);
    float*    bias0  = (float*)alloc(G_ * 4);
    float*    bias1  = (float*)alloc(G_ * 4);
    float*    dlsdf  = (float*)alloc((size_t)B_ * CTX_ * 4);
    float*    dlslo  = (float*)alloc((size_t)B_ * CTX_ * 4);
    float*    dlssc  = (float*)alloc((size_t)B_ * CTX_ * 4);
    float*    st_h0  = (float*)alloc(B_ * HID_ * 4);
    float*    st_c0  = (float*)alloc(B_ * HID_ * 4);
    float*    st_h1  = (float*)alloc(B_ * HID_ * 4);
    float*    st_c1  = (float*)alloc(B_ * HID_ * 4);
    float*    ysamp  = (float*)alloc((size_t)B_ * NS_ * (HOR_ - 1) * 4);

    k_prep<<<B_, 256, 0, stream>>>(X, mask, emb, tscale, buf0, feat);
    k_wcvt<<<G_, 64, 0, stream>>>(Wih0, Whh0, Wih1, Whh1, bih0, bhh0, bih1, bhh1,
                                  Wih0h, Whh0h, Wih1h, Whh1h, bias0, bias1);
    k_ctx<<<B_ / 16, 256, 0, stream>>>(feat, Wih0h, Whh0h, Wih1h, Whh1h, bias0, bias1,
                                       w_df, b_df, w_lo, b_lo, w_sc, b_sc,
                                       dlsdf, dlslo, dlssc, st_h0, st_c0, st_h1, st_c1);
    k_hor<<<(B_ * NS_) / 16, 256, 0, stream>>>(feat, Wih0h, Whh0h, Wih1h, Whh1h, bias0, bias1,
                                               w_df, b_df, w_lo, b_lo, w_sc, b_sc,
                                               dlsdf, dlslo, dlssc,
                                               st_h0, st_c0, st_h1, st_c1,
                                               tscale, buf0, ysamp);
    dim3 mgrid(TM1_, 2);
    k_median<<<mgrid, 128, 0, stream>>>(dlsdf, dlslo, dlssc, tscale, ysamp, (float*)d_out);
}
